// Indexer_43885975830881
// MI455X (gfx1250) — matrix operand order repaired and checked by reading, not compiled, not run on hardware
//
#include <hip/hip_runtime.h>
#include <hip/hip_bf16.h>
#include <stddef.h>

// ---------------- CDNA5 WMMA types ----------------
typedef __attribute__((ext_vector_type(16))) __bf16        v16bf;
typedef __attribute__((ext_vector_type(8)))  float         v8f;
typedef __attribute__((ext_vector_type(8)))  unsigned int  v8u;
typedef __attribute__((ext_vector_type(4)))  unsigned int  tdm4u;
typedef __attribute__((ext_vector_type(8)))  unsigned int  tdm8u;

#define S_LEN    4096
#define D_MODEL  2048
#define N_HEADS  16
#define HEAD_DIM 128
#define TOPK     2048

__device__ __forceinline__ unsigned short f2bf(float f) {
    unsigned u = __builtin_bit_cast(unsigned, f);
    u = (u + 0x7FFFu + ((u >> 16) & 1u)) >> 16;   // round-to-nearest-even
    return (unsigned short)u;
}

// A fragment: 16x32 bf16, lane lo = row M, hi = K half.
__device__ __forceinline__ v16bf frag_a16(const unsigned int* w, int hi) {
    v8u u;
#pragma unroll
    for (int j = 0; j < 8; ++j) u[j] = w[8 * hi + j];
    return __builtin_bit_cast(v16bf, u);
}
// B fragment: 32x16 bf16, lane lo = column N, hi = K half.
__device__ __forceinline__ v16bf frag_b16(const unsigned int* w, int hi) {
    v8u u;
#pragma unroll
    for (int v = 0; v < 8; ++v) u[v] = w[8 * hi + v];
    return __builtin_bit_cast(v16bf, u);
}

// ---------------- Tensor Data Mover: 2D bf16 tile -> LDS ----------------
// Builds D# groups 0/1 (2D tensor, data_size=2B) and issues tensor_load_to_lds.
// Issue from one wave only; caller must s_wait_tensorcnt + barrier.
__device__ __forceinline__ void tdm_load_2d_bf16(
    unsigned lds_off, const void* gptr,
    unsigned tensor_d0, unsigned tensor_d1,
    unsigned long long stride0_elems,
    unsigned tile_d0, unsigned tile_d1) {
    unsigned long long ga = (unsigned long long)(size_t)gptr;
    tdm4u g0;
    g0[0] = 1u;                                           // count=1, user mode
    g0[1] = lds_off;                                      // lds_addr (bytes)
    g0[2] = (unsigned)(ga & 0xFFFFFFFFull);               // global_addr[31:0]
    g0[3] = (unsigned)((ga >> 32) & 0x01FFFFFFull)        // global_addr[56:32]
            | 0x80000000u;                                // type=2 ("image")
    tdm8u g1;
    g1[0] = 0x10000u;                                     // data_size=1 -> 2 bytes
    g1[1] = (tensor_d0 & 0xFFFFu) << 16;                  // tensor_dim0[15:0]
    g1[2] = ((tensor_d0 >> 16) & 0xFFFFu)                 // tensor_dim0[31:16]
            | ((tensor_d1 & 0xFFFFu) << 16);              // tensor_dim1[15:0]
    g1[3] = ((tensor_d1 >> 16) & 0xFFFFu)                 // tensor_dim1[31:16]
            | ((tile_d0 & 0xFFFFu) << 16);                // tile_dim0
    g1[4] = (tile_d1 & 0xFFFFu);                          // tile_dim1 (tile_dim2=0)
    g1[5] = (unsigned)(stride0_elems & 0xFFFFFFFFull);    // tensor_dim0_stride lo
    g1[6] = (unsigned)((stride0_elems >> 32) & 0xFFFFull);// stride hi; dim1_stride=0
    g1[7] = 0u;
    asm volatile("tensor_load_to_lds %0, %1" :: "s"(g0), "s"(g1) : "memory");
}

// ---------------- fp32 -> bf16 convert ----------------
__global__ void cvt_f32_bf16(const float* __restrict__ src,
                             unsigned short* __restrict__ dst, int n) {
    int i = blockIdx.x * blockDim.x + threadIdx.x;
    int stride = gridDim.x * blockDim.x;
    for (; i < n; i += stride) dst[i] = f2bf(src[i]);
}

// ---------------- weights = x @ Ww * (N_HEADS*HEAD_DIM)^-0.5 ----------------
__global__ __launch_bounds__(256) void weights_kernel(
    const float* __restrict__ x, const float* __restrict__ Ww,
    float* __restrict__ wts) {
    int wave = (blockIdx.x * 256 + threadIdx.x) >> 5;
    int lane = threadIdx.x & 31;
    int s = wave >> 4;
    int n = wave & 15;
    if (s >= S_LEN) return;
    float sum = 0.f;
    for (int k = lane; k < D_MODEL; k += 32)
        sum += x[(size_t)s * D_MODEL + k] * Ww[(size_t)k * N_HEADS + n];
#pragma unroll
    for (int off = 16; off > 0; off >>= 1) sum += __shfl_down(sum, off, 32);
    if (lane == 0) wts[s * N_HEADS + n] = sum * 0.02209708691207961f; // 1/sqrt(2048)
}

// ---------------- projection + RMSNorm + RoPE (bf16 WMMA) ----------------
__global__ __launch_bounds__(256) void proj_norm_rope(
    const unsigned short* __restrict__ xb,   // S x 2048 bf16
    const unsigned short* __restrict__ wb,   // 2048 x ldN bf16
    const float* __restrict__ normw,         // 128
    const float* __restrict__ cosb,          // S x 128
    const float* __restrict__ sinb,          // S x 128
    unsigned short* __restrict__ outb,       // S x ldN bf16
    int ldN) {
    __shared__ unsigned int ldsA[16][16];
    __shared__ unsigned int ldsBt[128][17];
    __shared__ float ldsC[16][128];
    __shared__ float ldsPart[16][16];
    __shared__ float ldsScale[16];

    const int tid  = threadIdx.x;
    const int wave = tid >> 5, lane = tid & 31, lo = lane & 15, hi = lane >> 4;
    const int s0    = blockIdx.x * 16;
    const int head  = blockIdx.y;
    const int cbase = head * HEAD_DIM;

    v8f acc = {0.f, 0.f, 0.f, 0.f, 0.f, 0.f, 0.f, 0.f};

    for (int k0 = 0; k0 < D_MODEL; k0 += 32) {
        { // stage A tile: 16 x 32 bf16
            int r = tid >> 4, dw = tid & 15;
            const unsigned int* g = (const unsigned int*)xb;
            ldsA[r][dw] = g[((size_t)(s0 + r) * D_MODEL + k0) / 2 + dw];
            if (k0 + 32 < D_MODEL)
                __builtin_prefetch(&xb[(size_t)(s0 + r) * D_MODEL + k0 + 32]);
        }
        { // stage B tile transposed: ldsBt[c][k], k-contiguous
            int c = tid & 127, kg = tid >> 7;
            unsigned short* row = (unsigned short*)&ldsBt[c][0];
#pragma unroll
            for (int i = 0; i < 16; ++i) {
                int kk = kg * 16 + i;
                row[kk] = wb[(size_t)(k0 + kk) * ldN + cbase + c];
            }
        }
        __syncthreads();
        v16bf a = frag_a16(&ldsA[lo][0], hi);
        v16bf b = frag_b16(&ldsBt[wave * 16 + lo][0], hi);
        acc = __builtin_amdgcn_wmma_f32_16x16x32_bf16(
            false, a, false, b, (short)0, acc, false, false);
        __syncthreads();
    }

#pragma unroll
    for (int r = 0; r < 8; ++r) ldsC[r + 8 * hi][wave * 16 + lo] = acc[r];
    __syncthreads();

    {
        int r = tid >> 4, g = tid & 15;
        float ss = 0.f;
#pragma unroll
        for (int j = 0; j < 8; ++j) { float v = ldsC[r][g + 16 * j]; ss += v * v; }
        ldsPart[r][g] = ss;
    }
    __syncthreads();
    if (tid < 16) {
        float ss = 0.f;
#pragma unroll
        for (int g = 0; g < 16; ++g) ss += ldsPart[tid][g];
        ldsScale[tid] = rsqrtf(ss * (1.f / HEAD_DIM) + 1e-6f);
    }
    __syncthreads();

    {
        int r = tid >> 4, c16 = tid & 15;
        float scale = ldsScale[r];
        int s = s0 + r;
#pragma unroll
        for (int j = 0; j < 8; ++j) {
            int d  = c16 + 16 * j;
            int d2 = (d + 64) & 127;
            float vn  = ldsC[r][d]  * scale * normw[d];
            float vp  = ldsC[r][d2] * scale * normw[d2];
            float rot = (d < 64) ? -vp : vp;
            float o = vn * cosb[(size_t)s * HEAD_DIM + d] +
                      rot * sinb[(size_t)s * HEAD_DIM + d];
            outb[(size_t)s * ldN + cbase + d] = f2bf(o);
        }
    }
}

// ---------------- index_score: sum_n w[s,n]*relu(q[s,n]·k[t]) + causal mask ----
// Block: 16 queries x 128 keys. K tile staged once via TDM (32KB resident),
// Q staged 4 heads at a time via TDM (16KB), 16 WMMAs per stage per wave.
__global__ __launch_bounds__(256) void score_kernel(
    const unsigned short* __restrict__ qb,   // S x (16*128) bf16
    const unsigned short* __restrict__ kb,   // S x 128 bf16
    const float* __restrict__ wts,           // S x 16
    float* __restrict__ outScore) {          // S x S
    __shared__ unsigned int ldsK[128][64];   // 128 keys x 128 bf16  (32 KB)
    __shared__ unsigned int ldsQ[16][256];   // 16 rows x 4 heads x 128 bf16 (16 KB)
    __shared__ float ldsW[256];

    const int tid = threadIdx.x, wave = tid >> 5, lane = tid & 31;
    const int lo = lane & 15, hi = lane >> 4;
    const int s0 = blockIdx.x * 16;
    const int t0 = blockIdx.y * 128;

    if (tid < 32) {
        // K tile: 128 rows x 128 bf16, row stride 128 (contiguous region)
        tdm_load_2d_bf16((unsigned)(size_t)&ldsK[0][0],
                         kb + (size_t)t0 * HEAD_DIM,
                         HEAD_DIM, S_LEN, HEAD_DIM, HEAD_DIM, 128);
    }
    ldsW[tid] = wts[(size_t)(s0 + (tid >> 4)) * N_HEADS + (tid & 15)];

    v8f acc = {0.f, 0.f, 0.f, 0.f, 0.f, 0.f, 0.f, 0.f};

    for (int h4 = 0; h4 < 4; ++h4) {
        if (tid < 32) {
            // Q tile: 16 rows x 512 bf16 (4 contiguous heads), row stride 2048
            tdm_load_2d_bf16((unsigned)(size_t)&ldsQ[0][0],
                             qb + (size_t)s0 * (N_HEADS * HEAD_DIM) + h4 * 512,
                             N_HEADS * HEAD_DIM, S_LEN, N_HEADS * HEAD_DIM,
                             512, 16);
            __builtin_amdgcn_s_wait_tensorcnt(0);
        }
        __syncthreads();
#pragma unroll
        for (int hh = 0; hh < 4; ++hh) {
            const int n = h4 * 4 + hh;
            v8f sn = {0.f, 0.f, 0.f, 0.f, 0.f, 0.f, 0.f, 0.f};
#pragma unroll
            for (int ch = 0; ch < 4; ++ch) {
                v16bf a = frag_a16(&ldsQ[lo][hh * 64 + ch * 16], hi);
                v16bf b = frag_b16(&ldsK[wave * 16 + lo][ch * 16], hi);
                sn = __builtin_amdgcn_wmma_f32_16x16x32_bf16(
                    false, a, false, b, (short)0, sn, false, false);
            }
#pragma unroll
            for (int r = 0; r < 8; ++r) {
                float sv = sn[r];
                sv = sv > 0.f ? sv : 0.f;            // relu
                acc[r] += sv * ldsW[(r + 8 * hi) * 16 + n];
            }
        }
        __syncthreads();
    }

#pragma unroll
    for (int r = 0; r < 8; ++r) {
        int s = s0 + r + 8 * hi;
        int t = t0 + wave * 16 + lo;
        float v = acc[r];
        if (t > s) v = -__builtin_inff();            // causal mask
        outScore[(size_t)s * S_LEN + t] = v;
    }
}

// ---------------- top-k via stable descending bitonic sort in LDS -------------
__global__ __launch_bounds__(256) void topk_kernel(
    const float* __restrict__ score, int* __restrict__ outIdx) {
    __shared__ float sv[S_LEN];
    __shared__ int   si[S_LEN];
    const int s = blockIdx.x;
    const int tid = threadIdx.x;
    for (int j = tid; j < S_LEN; j += 256) {
        sv[j] = score[(size_t)s * S_LEN + j];
        si[j] = j;
    }
    __syncthreads();
    for (int k = 2; k <= S_LEN; k <<= 1) {
        for (int j = k >> 1; j > 0; j >>= 1) {
            for (int t = tid; t < S_LEN; t += 256) {
                int p = t ^ j;
                if (p > t) {
                    bool dir = ((t & k) == 0);
                    float va = sv[t], vb = sv[p];
                    int   ia = si[t], ib = si[p];
                    bool aFirst = (va > vb) || (va == vb && ia < ib);
                    if (aFirst != dir) {
                        sv[t] = vb; sv[p] = va;
                        si[t] = ib; si[p] = ia;
                    }
                }
            }
            __syncthreads();
        }
    }
    for (int j = tid; j < TOPK; j += 256)
        outIdx[(size_t)s * TOPK + j] = si[j];
}

// ---------------- launcher ----------------
extern "C" void kernel_launch(void* const* d_in, const int* in_sizes, int n_in,
                              void* d_out, int out_size, void* d_ws, size_t ws_size,
                              hipStream_t stream) {
    (void)in_sizes; (void)n_in; (void)out_size; (void)ws_size;
    const float* x    = (const float*)d_in[0];
    const float* Wq   = (const float*)d_in[1];
    const float* Wk   = (const float*)d_in[2];
    const float* Ww   = (const float*)d_in[3];
    const float* qn_w = (const float*)d_in[4];
    const float* kn_w = (const float*)d_in[5];
    const float* cosb = (const float*)d_in[6];
    const float* sinb = (const float*)d_in[7];

    int*   outIdx   = (int*)d_out;
    float* outScore = (float*)d_out + (size_t)S_LEN * TOPK;

    char* ws = (char*)d_ws;
    unsigned short* xb  = (unsigned short*)ws;                        // S*2048
    unsigned short* wqb = (unsigned short*)(ws + 16777216);           // 2048*2048
    unsigned short* wkb = (unsigned short*)(ws + 16777216 + 8388608); // 2048*128
    unsigned short* qb  = (unsigned short*)(ws + 25690112);           // S*2048
    unsigned short* kb  = (unsigned short*)(ws + 42467328);           // S*128
    float*          wts = (float*)(ws + 43515904);                    // S*16

    cvt_f32_bf16<<<4096, 256, 0, stream>>>(x,  xb,  S_LEN * D_MODEL);
    cvt_f32_bf16<<<4096, 256, 0, stream>>>(Wq, wqb, D_MODEL * N_HEADS * HEAD_DIM);
    cvt_f32_bf16<<<1024, 256, 0, stream>>>(Wk, wkb, D_MODEL * HEAD_DIM);

    weights_kernel<<<(S_LEN * N_HEADS) / 8, 256, 0, stream>>>(x, Ww, wts);

    proj_norm_rope<<<dim3(S_LEN / 16, N_HEADS), 256, 0, stream>>>(
        xb, wqb, qn_w, cosb, sinb, qb, N_HEADS * HEAD_DIM);
    proj_norm_rope<<<dim3(S_LEN / 16, 1), 256, 0, stream>>>(
        xb, wkb, kn_w, cosb, sinb, kb, HEAD_DIM);

    score_kernel<<<dim3(S_LEN / 16, S_LEN / 128), 256, 0, stream>>>(
        qb, kb, wts, outScore);

    topk_kernel<<<S_LEN, 256, 0, stream>>>(outScore, outIdx);
}
